// RWKV7_Block_32109175505560
// MI455X (gfx1250) — matrix operand order repaired and checked by reading, not compiled, not run on hardware
//
#include <hip/hip_runtime.h>

typedef __attribute__((ext_vector_type(16))) __bf16 v16bf;
typedef __attribute__((ext_vector_type(8)))  float  v8f;

#define TILE_M 128
#define TILE_N 128
#define TILE_K 32

#ifndef __has_builtin
#define __has_builtin(x) 0
#endif
#if __has_builtin(__builtin_amdgcn_global_load_async_to_lds_b128)
#define USE_ASYNC_LDS 1
#else
#define USE_ASYNC_LDS 0
#endif

// builtin prototype (from the round-2 diagnostic) takes pointers to a 16B int
// vector: AS1 (global / __device__) source, AS3 (LDS) destination.
typedef int b128_t __attribute__((vector_size(16)));
typedef __attribute__((address_space(1))) b128_t* gptr128;
typedef __attribute__((address_space(3))) b128_t* lptr128;

union FragB16 { unsigned u[8]; v16bf v; };
union Acc8    { v8f v; float f[8]; };

__device__ __forceinline__ float sigmoidf_(float x) { return 1.0f / (1.0f + __expf(-x)); }

__device__ __forceinline__ void wait_asynccnt0() {
#if __has_builtin(__builtin_amdgcn_s_wait_asynccnt)
    __builtin_amdgcn_s_wait_asynccnt(0);
#else
    asm volatile("s_wait_asynccnt 0x0" ::: "memory");
#endif
}

// ---------------------------------------------------------------------------
// f32 -> bf16 conversion (weights)
// ---------------------------------------------------------------------------
__global__ void cvt_f32_bf16(const float* __restrict__ in, __bf16* __restrict__ out, int n) {
    int i = blockIdx.x * blockDim.x + threadIdx.x;
    if (i < n) out[i] = (__bf16)in[i];
}

// ---------------------------------------------------------------------------
// block-wide sum over 256 threads (8 wave32s)
// ---------------------------------------------------------------------------
__device__ __forceinline__ float block_sum256(float v, float* sbuf) {
    #pragma unroll
    for (int o = 16; o > 0; o >>= 1) v += __shfl_xor(v, o, 32);
    __syncthreads();
    if ((threadIdx.x & 31) == 0) sbuf[threadIdx.x >> 5] = v;
    __syncthreads();
    float r = 0.0f;
    #pragma unroll
    for (int i = 0; i < 8; ++i) r += sbuf[i];
    return r;
}

// ---------------------------------------------------------------------------
// LayerNorm(x) + time-shift mix -> xm (bf16). One block per (b,t) row, D=1024.
// ---------------------------------------------------------------------------
__global__ __launch_bounds__(256) void ln1_shift_kernel(
    const float* __restrict__ x, const float* __restrict__ w, const float* __restrict__ b,
    __bf16* __restrict__ xm, int T, int D)
{
    __shared__ float sbuf[8];
    int row = blockIdx.x;
    int t   = row % T;
    const float* xc = x + (size_t)row * D;
    const float* xp = xc - D;  // valid only when t > 0

    float c[4], p[4];
    float sc = 0.f, sc2 = 0.f, sp = 0.f, sp2 = 0.f;
    #pragma unroll
    for (int q = 0; q < 4; ++q) {
        int ch = threadIdx.x + q * 256;
        float v = xc[ch]; c[q] = v; sc += v; sc2 += v * v;
        float u = (t > 0) ? xp[ch] : 0.f; p[q] = u; sp += u; sp2 += u * u;
    }
    float invD = 1.0f / (float)D;
    float mc  = block_sum256(sc,  sbuf) * invD;
    float mc2 = block_sum256(sc2, sbuf) * invD;
    float mp  = block_sum256(sp,  sbuf) * invD;
    float mp2 = block_sum256(sp2, sbuf) * invD;
    float rc = rsqrtf(mc2 - mc * mc + 1e-5f);
    float rp = rsqrtf(mp2 - mp * mp + 1e-5f);

    #pragma unroll
    for (int q = 0; q < 4; ++q) {
        int ch = threadIdx.x + q * 256;
        float ww = w[ch], bb = b[ch];
        float xxc = (c[q] - mc) * rc * ww + bb;
        float xxp = (t > 0) ? ((p[q] - mp) * rp * ww + bb) : 0.f;
        xm[(size_t)row * D + ch] = (__bf16)(0.5f * (xxc + xxp));
    }
}

// ---------------------------------------------------------------------------
// WMMA bf16 GEMM: C[M,N] = A[M,K] * Bw[N,K]^T (+ bias)
//   Double-buffered LDS; async global->LDS (ASYNCcnt) when available.
//   8 waves: wave (wm,wn) in 2x4 owns a 64x32 output region = 4x2 16x16 tiles.
//   epilogue 0: store f32.  epilogue 1: relu()^2 -> store bf16 (for k_ffn).
// ---------------------------------------------------------------------------
__global__ __launch_bounds__(256) void gemm_bf16_wmma(
    const __bf16* __restrict__ A,
    const __bf16* __restrict__ Bw,
    const float*  __restrict__ bias,
    float*        __restrict__ Cf,
    __bf16*       __restrict__ Cbf,
    int M, int N, int K, int epilogue)
{
    __shared__ __align__(16) __bf16 lA[2][TILE_M * TILE_K];
    __shared__ __align__(16) __bf16 lB[2][TILE_N * TILE_K];

    const int tid  = threadIdx.x;
    const int lane = tid & 31;
    const int wv   = tid >> 5;   // 0..7
    const int wm   = wv >> 2;    // 0..1  (64 rows each)
    const int wn   = wv & 3;     // 0..3  (32 cols each)
    const int rowBase = blockIdx.y * TILE_M;
    const int colBase = blockIdx.x * TILE_N;

    // this thread's two (row, 16B-chunk) slots of each 128x32 tile
    const int li0  = tid;        // 0..255
    const int li1  = tid + 256;  // 256..511
    const int row0 = li0 >> 2, c40 = li0 & 3;
    const int row1 = li1 >> 2, c41 = li1 & 3;

    auto issue_copy = [&](int kt, int buf) {
#if USE_ASYNC_LDS
        // one-instruction global->LDS DMA per 16B chunk, tracked by ASYNCcnt
        __builtin_amdgcn_global_load_async_to_lds_b128(
            (gptr128)(void*)(A  + (size_t)(rowBase + row0) * K + kt + c40 * 8),
            (lptr128)(void*)(&lA[buf][row0 * TILE_K + c40 * 8]), 0, 0);
        __builtin_amdgcn_global_load_async_to_lds_b128(
            (gptr128)(void*)(A  + (size_t)(rowBase + row1) * K + kt + c41 * 8),
            (lptr128)(void*)(&lA[buf][row1 * TILE_K + c41 * 8]), 0, 0);
        __builtin_amdgcn_global_load_async_to_lds_b128(
            (gptr128)(void*)(Bw + (size_t)(colBase + row0) * K + kt + c40 * 8),
            (lptr128)(void*)(&lB[buf][row0 * TILE_K + c40 * 8]), 0, 0);
        __builtin_amdgcn_global_load_async_to_lds_b128(
            (gptr128)(void*)(Bw + (size_t)(colBase + row1) * K + kt + c41 * 8),
            (lptr128)(void*)(&lB[buf][row1 * TILE_K + c41 * 8]), 0, 0);
#else
        // batched loads first so they pipeline, then LDS stores
        uint4 a0 = *(const uint4*)(A  + (size_t)(rowBase + row0) * K + kt + c40 * 8);
        uint4 a1 = *(const uint4*)(A  + (size_t)(rowBase + row1) * K + kt + c41 * 8);
        uint4 b0 = *(const uint4*)(Bw + (size_t)(colBase + row0) * K + kt + c40 * 8);
        uint4 b1 = *(const uint4*)(Bw + (size_t)(colBase + row1) * K + kt + c41 * 8);
        *(uint4*)(&lA[buf][row0 * TILE_K + c40 * 8]) = a0;
        *(uint4*)(&lA[buf][row1 * TILE_K + c41 * 8]) = a1;
        *(uint4*)(&lB[buf][row0 * TILE_K + c40 * 8]) = b0;
        *(uint4*)(&lB[buf][row1 * TILE_K + c41 * 8]) = b1;
        if (kt + TILE_K < K) {
            __builtin_prefetch(A  + (size_t)(rowBase + (tid >> 1)) * K + kt + TILE_K, 0, 0);
            __builtin_prefetch(Bw + (size_t)(colBase + (tid >> 1)) * K + kt + TILE_K, 0, 0);
        }
#endif
    };

    v8f acc[4][2] = {};

    const int mloc  = lane & 15;
    const int koffA = (lane >> 4) * 8;   // 16-bit A layout: lane half selects K+8
    const int koffB = (lane >> 4) * 8;   // 16-bit B layout: lane half selects K+8

    const int nk = K / TILE_K;
    issue_copy(0, 0);

    for (int ktile = 0; ktile < nk; ++ktile) {
        const int cur = ktile & 1;
#if USE_ASYNC_LDS
        wait_asynccnt0();
#endif
        __syncthreads();   // all copies into buf[cur] visible; all waves done reading buf[cur^1]

        if (ktile + 1 < nk) issue_copy((ktile + 1) * TILE_K, cur ^ 1);

        // gather fragments per ISA 7.12.2 16-bit layouts
        FragB16 afr[4], bfr[2];
        #pragma unroll
        for (int i = 0; i < 4; ++i) {
            const __bf16* rowp = &lA[cur][(wm * 64 + i * 16 + mloc) * TILE_K];
            #pragma unroll
            for (int vi = 0; vi < 8; ++vi) {
                int kb = ((vi < 4) ? (2 * vi) : (16 + 2 * (vi - 4))) + koffA;
                afr[i].u[vi] = *(const unsigned*)(rowp + kb);
            }
        }
        #pragma unroll
        for (int j = 0; j < 2; ++j) {
            const __bf16* rowp = &lB[cur][(wn * 32 + j * 16 + mloc) * TILE_K];
            #pragma unroll
            for (int vi = 0; vi < 8; ++vi)
                bfr[j].u[vi] = *(const unsigned*)(rowp + koffB + ((vi < 4) ? (2 * vi) : (16 + 2 * (vi - 4))));
        }

        #pragma unroll
        for (int i = 0; i < 4; ++i)
            #pragma unroll
            for (int j = 0; j < 2; ++j)
                acc[i][j] = __builtin_amdgcn_wmma_f32_16x16x32_bf16(
                    false, afr[i].v, false, bfr[j].v, (short)0, acc[i][j], false, false);
    }

    // epilogue: C/D layout — lane<16: M=e, N=lane; lane>=16: M=8+e, N=lane-16
    const int nloc  = lane & 15;
    const int mhalf = (lane >> 4) * 8;
    #pragma unroll
    for (int j = 0; j < 2; ++j) {
        int col = colBase + wn * 32 + j * 16 + nloc;
        float bv = bias ? bias[col] : 0.0f;
        #pragma unroll
        for (int i = 0; i < 4; ++i) {
            Acc8 a8; a8.v = acc[i][j];
            #pragma unroll
            for (int e = 0; e < 8; ++e) {
                int row = rowBase + wm * 64 + i * 16 + mhalf + e;
                float cv = a8.f[e] + bv;
                size_t idx = (size_t)row * N + col;
                if (epilogue == 1) { float tpos = cv > 0.f ? cv : 0.f; Cbf[idx] = (__bf16)(tpos * tpos); }
                else               { Cf[idx] = cv; }
            }
        }
    }
}

// ---------------------------------------------------------------------------
// kvab = k*v + a*b
// ---------------------------------------------------------------------------
__global__ void kvab_kernel(const float* __restrict__ k, const float* __restrict__ v,
                            const float* __restrict__ a, const float* __restrict__ b,
                            float* __restrict__ out, size_t n) {
    size_t i = (size_t)blockIdx.x * blockDim.x + threadIdx.x;
    if (i < n) out[i] = k[i] * v[i] + a[i] * b[i];
}

// ---------------------------------------------------------------------------
// per-(b,d) sequential decay scan over T; coalesced across d
// ---------------------------------------------------------------------------
__global__ void scan_kernel(const float* __restrict__ kvab, const float* __restrict__ st0,
                            const float* __restrict__ decay, float* __restrict__ states,
                            float* __restrict__ fstate, int B, int T, int D) {
    int idx = blockIdx.x * blockDim.x + threadIdx.x;
    if (idx >= B * D) return;
    int bb = idx / D, d = idx - bb * D;
    float w = __expf(-__expf(decay[d]));
    float s = st0[idx];
    const float* u = kvab + (size_t)bb * T * D + d;
    float* so = states + (size_t)bb * T * D + d;
    for (int t = 0; t < T; ++t) {
        s = s * w + u[(size_t)t * D];
        so[(size_t)t * D] = s;
    }
    fstate[idx] = s;
}

// ---------------------------------------------------------------------------
// GroupNorm(states, G=32) + sigmoid(r)*gn*sigmoid(g) + residual, then LN2 -> bf16
// One block per row; each 32-channel group lives exactly in one wave32.
// ---------------------------------------------------------------------------
__global__ __launch_bounds__(256) void gn_mix_ln2_kernel(
    const float* __restrict__ states, const float* __restrict__ rr, const float* __restrict__ gg,
    const float* __restrict__ x,
    const float* __restrict__ snw, const float* __restrict__ snb,
    const float* __restrict__ ln2w, const float* __restrict__ ln2b,
    float* __restrict__ x1, __bf16* __restrict__ xx2, int D)
{
    __shared__ float sbuf[8];
    size_t base = (size_t)blockIdx.x * D;
    float xv[4];
    float s1 = 0.f, s2 = 0.f;
    #pragma unroll
    for (int q = 0; q < 4; ++q) {
        int ch = threadIdx.x + q * 256;
        float s = states[base + ch];
        float sum = s, sq = s * s;
        #pragma unroll
        for (int o = 16; o > 0; o >>= 1) { sum += __shfl_xor(sum, o, 32); sq += __shfl_xor(sq, o, 32); }
        float mu  = sum * (1.f / 32.f);
        float var = sq  * (1.f / 32.f) - mu * mu;
        float gn  = (s - mu) * rsqrtf(var + 1e-5f) * snw[ch] + snb[ch];
        float mixed = sigmoidf_(rr[base + ch]) * gn * sigmoidf_(gg[base + ch]);
        float xo = x[base + ch] + mixed;
        x1[base + ch] = xo;
        xv[q] = xo;
        s1 += xo; s2 += xo * xo;
    }
    float invD = 1.0f / (float)D;
    float mu2 = block_sum256(s1, sbuf) * invD;
    float m2  = block_sum256(s2, sbuf) * invD;
    float rinv = rsqrtf(m2 - mu2 * mu2 + 1e-5f);
    #pragma unroll
    for (int q = 0; q < 4; ++q) {
        int ch = threadIdx.x + q * 256;
        xx2[base + ch] = (__bf16)((xv[q] - mu2) * rinv * ln2w[ch] + ln2b[ch]);
    }
}

// ---------------------------------------------------------------------------
// out = x1 + sigmoid(ffn_r_out) * ffn_v_out
// ---------------------------------------------------------------------------
__global__ void ffn_fuse_kernel(const float* __restrict__ x1, const float* __restrict__ rr,
                                const float* __restrict__ vv, float* __restrict__ out, size_t n) {
    size_t i = (size_t)blockIdx.x * blockDim.x + threadIdx.x;
    if (i < n) out[i] = x1[i] + sigmoidf_(rr[i]) * vv[i];
}

// ---------------------------------------------------------------------------
extern "C" void kernel_launch(void* const* d_in, const int* in_sizes, int n_in,
                              void* d_out, int out_size, void* d_ws, size_t ws_size,
                              hipStream_t stream) {
    (void)n_in; (void)out_size; (void)ws_size;
    const float* x    = (const float*)d_in[0];
    const float* st0  = (const float*)d_in[1];
    const float* ln1w = (const float*)d_in[2];
    const float* ln1b = (const float*)d_in[3];
    const float* ln2w = (const float*)d_in[4];
    const float* ln2b = (const float*)d_in[5];
    const float* snw  = (const float*)d_in[6];
    const float* snb  = (const float*)d_in[7];
    const float* Wr   = (const float*)d_in[8];
    const float* Wg   = (const float*)d_in[9];
    const float* Wk   = (const float*)d_in[10];
    const float* Wv   = (const float*)d_in[11];
    const float* Wa   = (const float*)d_in[12];
    const float* Wb   = (const float*)d_in[13];
    const float* decay= (const float*)d_in[14];
    const float* fkw  = (const float*)d_in[15];
    const float* fkb  = (const float*)d_in[16];
    const float* fvw  = (const float*)d_in[17];
    const float* fvb  = (const float*)d_in[18];
    const float* frw  = (const float*)d_in[19];
    const float* frb  = (const float*)d_in[20];

    const int D  = in_sizes[2];
    const int B  = in_sizes[1] / D;
    const int BT = in_sizes[0] / D;
    const int T  = BT / B;
    const int H  = in_sizes[16];
    const size_t NE = (size_t)BT * D;

    char* ws = (char*)d_ws;
    size_t off = 0;
    auto alloc = [&](size_t bytes) -> void* {
        void* p = ws + off;
        off += (bytes + 255) & ~(size_t)255;
        return p;
    };

    const float* wsrc[9] = {Wr, Wg, Wk, Wv, Wa, Wb, fkw, fvw, frw};
    int wcnt[9] = {D*D, D*D, D*D, D*D, D*D, D*D, H*D, D*H, D*D};
    __bf16* wbf[9];
    for (int i = 0; i < 9; ++i) wbf[i] = (__bf16*)alloc((size_t)wcnt[i] * 2);

    __bf16* xm   = (__bf16*)alloc(NE * 2);
    float*  bufR = (float*)alloc(NE * 4);
    float*  bufG = (float*)alloc(NE * 4);
    float*  bufK = (float*)alloc(NE * 4);
    float*  bufV = (float*)alloc(NE * 4);
    float*  bufA = (float*)alloc(NE * 4);
    float*  bufB = (float*)alloc(NE * 4);

    // buffer reuse after each producer's last consumer:
    float*  kvab   = bufK;                               // k dead after kvab
    float*  states = bufV;                               // v dead after kvab
    float*  x1     = bufA;                               // a dead after kvab
    __bf16* xx2    = (__bf16*)bufB;                      // b dead after kvab (32MB)
    __bf16* kffn   = (__bf16*)((char*)bufB + NE * 2);    // second half of bufB
    float*  rrout  = bufR;                               // r dead after gn kernel
    float*  vvout  = bufG;                               // g dead after gn kernel

    // 1) weights -> bf16
    for (int i = 0; i < 9; ++i)
        cvt_f32_bf16<<<dim3((wcnt[i] + 255) / 256), dim3(256), 0, stream>>>(wsrc[i], wbf[i], wcnt[i]);

    // 2) LN1 + time-shift -> xm (bf16)
    ln1_shift_kernel<<<dim3(BT), dim3(256), 0, stream>>>(x, ln1w, ln1b, xm, T, D);

    // 3) six projections (WMMA bf16)
    dim3 gemm_blk(256);
    dim3 gDD(D / TILE_N, BT / TILE_M);
    gemm_bf16_wmma<<<gDD, gemm_blk, 0, stream>>>(xm, wbf[0], nullptr, bufR, nullptr, BT, D, D, 0); // r
    gemm_bf16_wmma<<<gDD, gemm_blk, 0, stream>>>(xm, wbf[1], nullptr, bufG, nullptr, BT, D, D, 0); // g
    gemm_bf16_wmma<<<gDD, gemm_blk, 0, stream>>>(xm, wbf[2], nullptr, bufK, nullptr, BT, D, D, 0); // k
    gemm_bf16_wmma<<<gDD, gemm_blk, 0, stream>>>(xm, wbf[3], nullptr, bufV, nullptr, BT, D, D, 0); // v
    gemm_bf16_wmma<<<gDD, gemm_blk, 0, stream>>>(xm, wbf[4], nullptr, bufA, nullptr, BT, D, D, 0); // a
    gemm_bf16_wmma<<<gDD, gemm_blk, 0, stream>>>(xm, wbf[5], nullptr, bufB, nullptr, BT, D, D, 0); // b

    // 4) kvab = k*v + a*b
    kvab_kernel<<<dim3((unsigned)((NE + 255) / 256)), dim3(256), 0, stream>>>(bufK, bufV, bufA, bufB, kvab, NE);

    // 5) decay scan; final state straight into d_out tail
    float* fstate = (float*)d_out + NE;
    scan_kernel<<<dim3((B * D + 255) / 256), dim3(256), 0, stream>>>(kvab, st0, decay, states, fstate, B, T, D);

    // 6) GroupNorm + mix + residual + LN2 -> xx2 (bf16)
    gn_mix_ln2_kernel<<<dim3(BT), dim3(256), 0, stream>>>(states, bufR, bufG, x, snw, snb, ln2w, ln2b, x1, xx2, D);

    // 7) FFN GEMMs
    dim3 gDH(H / TILE_N, BT / TILE_M);
    gemm_bf16_wmma<<<gDH, gemm_blk, 0, stream>>>(xx2,  wbf[6], fkb, nullptr, kffn, BT, H, D, 1); // relu^2 -> bf16
    gemm_bf16_wmma<<<gDD, gemm_blk, 0, stream>>>(xx2,  wbf[8], frb, rrout, nullptr, BT, D, D, 0);
    gemm_bf16_wmma<<<gDD, gemm_blk, 0, stream>>>(kffn, wbf[7], fvb, vvout, nullptr, BT, D, H, 0);

    // 8) out = x1 + sigmoid(rr) * vv
    ffn_fuse_kernel<<<dim3((unsigned)((NE + 255) / 256)), dim3(256), 0, stream>>>(x1, rrout, vvout, (float*)d_out, NE);
}
